// MultiHeadAttention_30855045055232
// MI455X (gfx1250) — matrix operand order repaired and checked by reading, not compiled, not run on hardware
//
#include <hip/hip_runtime.h>

#define D_MODEL 1024
#define N_HEADS 16
#define HEAD_DIM 64
#define BATCH 2
#define SEQ 2048
#define MROWS (BATCH * SEQ) /* 4096 */

typedef __attribute__((ext_vector_type(16))) __bf16 v16bf;
typedef __attribute__((ext_vector_type(8)))  float  v8f;

union AFrag { v16bf v; unsigned int u[8]; };

__device__ __forceinline__ unsigned short f32_to_bf16(float f) {
    unsigned int u = __float_as_uint(f);
    u += 0x7fffu + ((u >> 16) & 1u);       // round-to-nearest-even
    return (unsigned short)(u >> 16);
}

// gfx1250 async global->LDS copy, 16B per lane. LDS dest is the low 32 bits of
// the generic pointer (generic LDS addr = aperture | lds_offset, ISA 10.2).
__device__ __forceinline__ void async_b128(const void* gsrc, unsigned lds_byte_addr) {
    asm volatile("global_load_async_to_lds_b128 %0, %1, off"
                 :: "v"(lds_byte_addr), "v"((unsigned long long)(size_t)gsrc)
                 : "memory");
}

__device__ __forceinline__ void wait_async0() {
#if __has_builtin(__builtin_amdgcn_s_wait_asynccnt)
    __builtin_amdgcn_s_wait_asynccnt(0);
#else
    asm volatile("s_wait_asynccnt 0x0" ::: "memory");
#endif
}

// ---------------------------------------------------------------- converts
__global__ __launch_bounds__(256)
void cvt_f32_bf16(const float* __restrict__ x, unsigned short* __restrict__ y, int n) {
    int i = (blockIdx.x * 256 + threadIdx.x) * 4;
    if (i + 3 < n) {
        float4 f = *(const float4*)(x + i);
        unsigned int lo = (unsigned int)f32_to_bf16(f.x) | ((unsigned int)f32_to_bf16(f.y) << 16);
        unsigned int hi = (unsigned int)f32_to_bf16(f.z) | ((unsigned int)f32_to_bf16(f.w) << 16);
        *(uint2*)(y + i) = make_uint2(lo, hi);
    }
}

// Pack W[K,N] f32 into a fragment-ready tiled bf16-pair layout:
//   Wt[((nBlk*(K/32) + kBlk)*128 + col)*16 + pair]  (pair = {W[2p][n],W[2p+1][n]})
// so a GEMM B tile (32k x 128n) is one contiguous 8KB block -> pure async copy.
__global__ __launch_bounds__(256)
void pack_w_tiles(const float* __restrict__ W, unsigned int* __restrict__ Wt, int K, int N) {
    int idx = blockIdx.x * 256 + threadIdx.x;
    if (idx < (K / 2) * N) {
        int p = idx / N, n = idx % N;
        unsigned int lo = f32_to_bf16(W[(2 * p) * N + n]);
        unsigned int hi = f32_to_bf16(W[(2 * p + 1) * N + n]);
        int kBlk = p >> 4, pr = p & 15;
        int nBlk = n >> 7, col = n & 127;
        size_t dst = (((size_t)nBlk * (K >> 5) + kBlk) * 128 + col) * 16 + pr;
        Wt[dst] = lo | (hi << 16);
    }
}

// ---------------------------------------------------------------- GEMM
// C[M,N] = A[M,K](bf16) * B(tiled pair-packed) + bias; out bf16 and/or f32.
// 128x128x32 block tile, 8 waves (4x2), wave tile 32x64 = 2x4 WMMA accumulators.
// Double-buffered async global->LDS staging; LDS rows padded to 20 dwords
// (16B aligned, 16 distinct bank groups -> conflict-free ds_load_b128).
#define LDS_PITCH 20
__global__ __launch_bounds__(256)
void gemm_bf16_wmma(const unsigned short* __restrict__ A,
                    const unsigned int* __restrict__ Bt,
                    const float* __restrict__ bias,
                    unsigned short* __restrict__ Cbf,
                    float* __restrict__ Cf,
                    int M, int N, int K) {
    __shared__ unsigned int Abuf[2][128 * LDS_PITCH];   // 10 KB each
    __shared__ unsigned int Bbuf[2][128 * LDS_PITCH];   // 10 KB each

    const int tid  = threadIdx.x;
    const int lane = tid & 31;
    const int wave = tid >> 5;
    const int wm   = wave >> 1;            // 0..3
    const int wn   = wave & 1;             // 0..1
    const int half = lane >> 4;
    const int l16  = lane & 15;

    const int bm = blockIdx.y * 128;
    const int bn = blockIdx.x * 128;
    const int Kp = K >> 1;
    const int kTiles = K >> 5;

    const unsigned int* A32 = (const unsigned int*)A;

    v8f acc[2][4] = {};

    const int srow = tid >> 1;             // 0..127 (A row / B col)
    const int scol = (tid & 1) * 8;        // dword chunk

    auto stage = [&](int buf, int k0) {
        const int kp0 = k0 >> 1;
        const unsigned int* ga = A32 + (size_t)(bm + srow) * Kp + kp0 + scol;
        unsigned la = (unsigned)(size_t)&Abuf[buf][srow * LDS_PITCH + scol];
        async_b128(ga, la);
        async_b128(ga + 4, la + 16);
        const unsigned int* gb =
            Bt + ((((size_t)(bn >> 7)) * kTiles + (k0 >> 5)) * 128 + srow) * 16 + scol;
        unsigned lb = (unsigned)(size_t)&Bbuf[buf][srow * LDS_PITCH + scol];
        async_b128(gb, lb);
        async_b128(gb + 4, lb + 16);
    };

    stage(0, 0);
    for (int k0 = 0; k0 < K; k0 += 32) {
        const int cur = (k0 >> 5) & 1;
        wait_async0();          // my copies into buf[cur] are in LDS
        __syncthreads();        // everyone's copies landed; prev compute done
        if (k0 + 32 < K) stage(cur ^ 1, k0 + 32);

        const unsigned int* Ab = &Abuf[cur][0];
        const unsigned int* Bb = &Bbuf[cur][0];

        AFrag a[2], b[4];
        #pragma unroll
        for (int m = 0; m < 2; ++m) {
            const int row = wm * 32 + m * 16 + l16;
            #pragma unroll
            for (int j = 0; j < 4; ++j) {
                a[m].u[j]     = Ab[row * LDS_PITCH + half * 4 + j];
                a[m].u[4 + j] = Ab[row * LDS_PITCH + 8 + half * 4 + j];
            }
        }
        #pragma unroll
        for (int n = 0; n < 4; ++n) {
            const int col = wn * 64 + n * 16 + l16;
            #pragma unroll
            for (int j = 0; j < 8; ++j) b[n].u[j] = Bb[col * LDS_PITCH + (j >> 2) * 8 + half * 4 + (j & 3)];
        }
        #pragma unroll
        for (int m = 0; m < 2; ++m)
            #pragma unroll
            for (int n = 0; n < 4; ++n)
                acc[m][n] = __builtin_amdgcn_wmma_f32_16x16x32_bf16(
                    false, a[m].v, false, b[n].v, (short)0, acc[m][n], false, false);
    }

    #pragma unroll
    for (int n = 0; n < 4; ++n) {
        const int col = bn + wn * 64 + n * 16 + l16;
        const float bb = bias ? bias[col] : 0.f;
        #pragma unroll
        for (int m = 0; m < 2; ++m) {
            const int row0 = bm + wm * 32 + m * 16 + half * 8;
            #pragma unroll
            for (int r = 0; r < 8; ++r) {
                float vv = acc[m][n][r] + bb;
                if (Cf)  Cf[(size_t)(row0 + r) * N + col] = vv;
                if (Cbf) Cbf[(size_t)(row0 + r) * N + col] = f32_to_bf16(vv);
            }
        }
    }
}

// ---------------------------------------------------------------- attention
// Flash attention, one (b,h) x 64-query tile per block. 4 waves, 16 q-rows each.
// Q/K/V layout: bf16 [B*S, D_MODEL] (head h at column h*64).
// K tile staged via async b128; V transposed manually; padded LDS rows.
__global__ __launch_bounds__(128)
void attn_wmma(const unsigned short* __restrict__ Qb,
               const unsigned short* __restrict__ Kb,
               const unsigned short* __restrict__ Vb,
               unsigned short* __restrict__ Ctx) {
    __shared__ unsigned int   Ks[64][36];      // [key][hd-pair], pad 32->36
    __shared__ unsigned short Vs[64][72];      // [hd][key] transposed, pad 64->72
    __shared__ unsigned short Ps[4][16][72];   // per-wave P tile, pad 64->72

    const int tid  = threadIdx.x;
    const int lane = tid & 31;
    const int wave = tid >> 5;
    const int half = lane >> 4;
    const int l16  = lane & 15;

    const int bh = blockIdx.y;
    const int b  = bh >> 4;
    const int h  = bh & 15;
    const int q0 = blockIdx.x * 64 + wave * 16;
    const int baseRow = b * SEQ;
    const int hoff = h * HEAD_DIM;
    const float scale = 0.125f;  // 1/sqrt(64)

    // Q fragments (two 16x32 K-chunks over Hd=64)
    AFrag aq[2];
    {
        const unsigned int* qrow =
            (const unsigned int*)(Qb + (size_t)(baseRow + q0 + l16) * D_MODEL + hoff);
        #pragma unroll
        for (int ks = 0; ks < 2; ++ks)
            #pragma unroll
            for (int j = 0; j < 4; ++j) {
                aq[ks].u[j]     = qrow[ks * 16 + half * 4 + j];
                aq[ks].u[4 + j] = qrow[ks * 16 + 8 + half * 4 + j];
            }
    }

    v8f o[4] = {};
    float mrow[8], lrow[8];
    #pragma unroll
    for (int r = 0; r < 8; ++r) { mrow[r] = -3.0e38f; lrow[r] = 0.f; }

    for (int j0 = 0; j0 < SEQ; j0 += 64) {
        __syncthreads();   // previous iteration's reads of Ks/Vs are done
        {
            const int key = tid >> 1;
            // K: async copy 64B (4 x b128), pair-packed rows
            const int pc = (tid & 1) * 16;
            const unsigned int* kr =
                (const unsigned int*)(Kb + (size_t)(baseRow + j0 + key) * D_MODEL + hoff) + pc;
            #pragma unroll
            for (int c = 0; c < 4; ++c)
                async_b128(kr + 4 * c, (unsigned)(size_t)&Ks[key][pc + 4 * c]);

            // V: transpose into [hd][key] (u16 scatter)
            const int hd0 = (tid & 1) * 32;
            const uint4* vg = (const uint4*)(Vb + (size_t)(baseRow + j0 + key) * D_MODEL + hoff + hd0);
            #pragma unroll
            for (int c = 0; c < 4; ++c) {
                uint4 vv = vg[c];
                unsigned int w[4] = { vv.x, vv.y, vv.z, vv.w };
                #pragma unroll
                for (int e = 0; e < 4; ++e) {
                    Vs[hd0 + c * 8 + e * 2 + 0][key] = (unsigned short)(w[e] & 0xffffu);
                    Vs[hd0 + c * 8 + e * 2 + 1][key] = (unsigned short)(w[e] >> 16);
                }
            }
        }
        wait_async0();
        __syncthreads();

        // scores = Q @ K^T (16 x 64 per wave)
        v8f s[4] = {};
        #pragma unroll
        for (int n = 0; n < 4; ++n) {
            #pragma unroll
            for (int ks = 0; ks < 2; ++ks) {
                AFrag bk;
                const int key = n * 16 + l16;
                #pragma unroll
                for (int j = 0; j < 8; ++j) bk.u[j] = Ks[key][ks * 16 + (j >> 2) * 8 + half * 4 + (j & 3)];
                s[n] = __builtin_amdgcn_wmma_f32_16x16x32_bf16(
                    false, aq[ks].v, false, bk.v, (short)0, s[n], false, false);
            }
        }

        // online softmax; row = half*8 + r lives in this lane's 16-group
        #pragma unroll
        for (int r = 0; r < 8; ++r) {
            float mx = -3.0e38f;
            #pragma unroll
            for (int n = 0; n < 4; ++n) mx = fmaxf(mx, s[n][r]);
            mx *= scale;
            #pragma unroll
            for (int off = 8; off >= 1; off >>= 1) mx = fmaxf(mx, __shfl_xor(mx, off, 32));
            float mnew  = fmaxf(mrow[r], mx);
            float alpha = __expf(mrow[r] - mnew);
            mrow[r] = mnew;
            float rsum = 0.f;
            #pragma unroll
            for (int n = 0; n < 4; ++n) {
                float p = __expf(s[n][r] * scale - mnew);
                rsum += p;
                Ps[wave][half * 8 + r][n * 16 + l16] = f32_to_bf16(p);
            }
            #pragma unroll
            for (int off = 8; off >= 1; off >>= 1) rsum += __shfl_xor(rsum, off, 32);
            lrow[r] = lrow[r] * alpha + rsum;
            #pragma unroll
            for (int n = 0; n < 4; ++n) o[n][r] *= alpha;
        }

        // ctx += P @ V (P re-fetched from per-wave LDS; DS in-order per wave)
        const unsigned int* prow = (const unsigned int*)&Ps[wave][l16][0];
        #pragma unroll
        for (int ks = 0; ks < 2; ++ks) {
            AFrag ap;
            #pragma unroll
            for (int j = 0; j < 4; ++j) {
                ap.u[j]     = prow[ks * 16 + half * 4 + j];
                ap.u[4 + j] = prow[ks * 16 + 8 + half * 4 + j];
            }
            #pragma unroll
            for (int n = 0; n < 4; ++n) {
                AFrag bvf;
                const unsigned int* vcol = (const unsigned int*)&Vs[n * 16 + l16][0];
                #pragma unroll
                for (int j = 0; j < 8; ++j) bvf.u[j] = vcol[ks * 16 + (j >> 2) * 8 + half * 4 + (j & 3)];
                o[n] = __builtin_amdgcn_wmma_f32_16x16x32_bf16(
                    false, ap.v, false, bvf.v, (short)0, o[n], false, false);
            }
        }
    }

    #pragma unroll
    for (int r = 0; r < 8; ++r) {
        float inv = 1.f / lrow[r];
        const size_t row = (size_t)(baseRow + q0 + half * 8 + r) * D_MODEL + hoff;
        #pragma unroll
        for (int n = 0; n < 4; ++n)
            Ctx[row + n * 16 + l16] = f32_to_bf16(o[n][r] * inv);
    }
}

// ---------------------------------------------------------------- launch
extern "C" void kernel_launch(void* const* d_in, const int* in_sizes, int n_in,
                              void* d_out, int out_size, void* d_ws, size_t ws_size,
                              hipStream_t stream) {
    const float* q  = (const float*)d_in[0];
    const float* k  = (const float*)d_in[1];
    const float* v  = (const float*)d_in[2];
    const float* Wq = (const float*)d_in[3];
    const float* bq = (const float*)d_in[4];
    const float* Wk = (const float*)d_in[5];
    const float* bk = (const float*)d_in[6];
    const float* Wv = (const float*)d_in[7];
    const float* bv = (const float*)d_in[8];
    const float* Wo = (const float*)d_in[9];
    const float* bo = (const float*)d_in[10];
    float* out = (float*)d_out;

    const size_t MD   = (size_t)MROWS * D_MODEL;                                 // 4M elems
    const size_t bfSz = MD * sizeof(unsigned short);                             // 8 MB
    const size_t wpSz = (size_t)(D_MODEL / 2) * D_MODEL * sizeof(unsigned int);  // 2 MB

    char* ws = (char*)d_ws;
    unsigned short* qb  = (unsigned short*)ws; ws += bfSz;
    unsigned short* kb  = (unsigned short*)ws; ws += bfSz;
    unsigned short* vb  = (unsigned short*)ws; ws += bfSz;
    unsigned int*   Wqp = (unsigned int*)ws;   ws += wpSz;
    unsigned int*   Wkp = (unsigned int*)ws;   ws += wpSz;
    unsigned int*   Wvp = (unsigned int*)ws;   ws += wpSz;
    unsigned int*   Wop = (unsigned int*)ws;   ws += wpSz;
    unsigned short* Qp  = (unsigned short*)ws; ws += bfSz;
    unsigned short* Kp  = (unsigned short*)ws; ws += bfSz;
    unsigned short* Vp  = (unsigned short*)ws; ws += bfSz;
    unsigned short* Ctx = qb;   // reuse: qb is dead after the Q projection

    const int cvtBlocks = (int)(MD / (256 * 4));
    cvt_f32_bf16<<<cvtBlocks, 256, 0, stream>>>(q, qb, (int)MD);
    cvt_f32_bf16<<<cvtBlocks, 256, 0, stream>>>(k, kb, (int)MD);
    cvt_f32_bf16<<<cvtBlocks, 256, 0, stream>>>(v, vb, (int)MD);

    const int pwBlocks = (D_MODEL / 2) * D_MODEL / 256;
    pack_w_tiles<<<pwBlocks, 256, 0, stream>>>(Wq, Wqp, D_MODEL, D_MODEL);
    pack_w_tiles<<<pwBlocks, 256, 0, stream>>>(Wk, Wkp, D_MODEL, D_MODEL);
    pack_w_tiles<<<pwBlocks, 256, 0, stream>>>(Wv, Wvp, D_MODEL, D_MODEL);
    pack_w_tiles<<<pwBlocks, 256, 0, stream>>>(Wo, Wop, D_MODEL, D_MODEL);

    dim3 gg(D_MODEL / 128, MROWS / 128);   // (8, 32)
    gemm_bf16_wmma<<<gg, 256, 0, stream>>>(qb, Wqp, bq, Qp, nullptr, MROWS, D_MODEL, D_MODEL);
    gemm_bf16_wmma<<<gg, 256, 0, stream>>>(kb, Wkp, bk, Kp, nullptr, MROWS, D_MODEL, D_MODEL);
    gemm_bf16_wmma<<<gg, 256, 0, stream>>>(vb, Wvp, bv, Vp, nullptr, MROWS, D_MODEL, D_MODEL);

    dim3 ag(SEQ / 64, BATCH * N_HEADS);    // (32, 32)
    attn_wmma<<<ag, 128, 0, stream>>>(Qp, Kp, Vp, Ctx);

    gemm_bf16_wmma<<<gg, 256, 0, stream>>>(Ctx, Wop, bo, nullptr, out, MROWS, D_MODEL, D_MODEL);
}
